// Controller_tran_43542378447525
// MI455X (gfx1250) — matrix operand order repaired and checked by reading, not compiled, not run on hardware
//
#include <hip/hip_runtime.h>
#include <math.h>

// ---- problem constants (from the reference) ----
#define NF        256     // NUM_FEATURE
#define NOPX      2048    // NUM_OP
#define NG4       8192    // 4 * NUM_OP
#define MAX_ORD   8       // MAX_ORDER
#define NBATCH    32
#define NACT      2048    // NUM_ACTION
#define REGC      1e-3f

#define KCH 64            // K-chunk per staged LDS tile (2 WMMA K-slices)
#define NIT (NOPX / KCH)  // 32 iterations (even; loop is unrolled by 2)

typedef __attribute__((ext_vector_type(16))) _Float16 v16h;
typedef __attribute__((ext_vector_type(8)))  _Float16 v8h;
typedef __attribute__((ext_vector_type(8)))  float    v8f;

// Async global->LDS staging (GLOBAL_LOAD_ASYNC_TO_LDS_B128, ASYNCcnt-tracked).
// The builtin takes int4-typed pointers: (AS1 int4* src, AS3 int4* dst, imm, imm).
#if defined(__gfx1250__) && __has_builtin(__builtin_amdgcn_global_load_async_to_lds_b128)
#define USE_ASYNC_LDS 1
typedef int v4i_vs __attribute__((vector_size(16)));
typedef __attribute__((address_space(1))) v4i_vs* agp_t;   // global int4*
typedef __attribute__((address_space(3))) v4i_vs* alp_t;   // LDS int4*
#else
#define USE_ASYNC_LDS 0
#endif

__device__ __forceinline__ void wait_async_zero() {
#if defined(__gfx1250__)
#if __has_builtin(__builtin_amdgcn_s_wait_asynccnt)
    __builtin_amdgcn_s_wait_asynccnt(0);
#else
    asm volatile("s_wait_asynccnt 0x0" ::: "memory");
#endif
#endif
}

__device__ __forceinline__ float fsig(float x) {
    return 1.0f / (1.0f + __expf(-x));
}
__device__ __forceinline__ float ftanh(float x) {
    x = fminf(15.0f, fmaxf(-15.0f, x));
    float e = __expf(2.0f * x);
    return (e - 1.0f) / (e + 1.0f);
}

// ---------------------------------------------------------------------------
// prep: bias = b_ih + b_hh ; X0 = f16(input0) ; C = 0 ; regbuf[block] = partial
// sum of W_ih[:256]^2 (first NF*NOPX elements of W_ih, fixed grid of 512).
// ---------------------------------------------------------------------------
__global__ __launch_bounds__(256) void prep_kernel(
    const float* __restrict__ input0, const float* __restrict__ W_ih,
    const float* __restrict__ b_ih,  const float* __restrict__ b_hh,
    float* __restrict__ bias, _Float16* __restrict__ X0,
    float* __restrict__ C, float* __restrict__ regbuf)
{
    const int stride = gridDim.x * blockDim.x;
    float reg = 0.0f;
    for (int i = blockIdx.x * blockDim.x + threadIdx.x; i < NF * NOPX; i += stride) {
        float x0 = input0[i];
        X0[i] = (_Float16)x0;
        C[i]  = 0.0f;
        float w = W_ih[i];             // rows [0,256) of W_ih == first NF*NOPX elems
        reg += w * w;
        if (i < NG4) bias[i] = b_ih[i] + b_hh[i];
    }
    __shared__ float red[256];
    red[threadIdx.x] = reg; __syncthreads();
    for (int s = 128; s > 0; s >>= 1) {
        if (threadIdx.x < s) red[threadIdx.x] += red[threadIdx.x + s];
        __syncthreads();
    }
    if (threadIdx.x == 0) regbuf[blockIdx.x] = red[0];
}

// ---------------------------------------------------------------------------
// convert: Wih16 = f16(W_ih) ; Wsum16 = f16(W_ih + W_hh)
// (for t>=1, x==h so gates = h @ (W_ih+W_hh)^T)
// ---------------------------------------------------------------------------
__global__ __launch_bounds__(256) void convert_kernel(
    const float* __restrict__ W_ih, const float* __restrict__ W_hh,
    _Float16* __restrict__ Wih16, _Float16* __restrict__ Wsum16)
{
    const size_t stride = (size_t)gridDim.x * blockDim.x;
    for (size_t i = (size_t)blockIdx.x * blockDim.x + threadIdx.x;
         i < (size_t)NG4 * NOPX; i += stride) {
        float a = W_ih[i], b = W_hh[i];
        Wih16[i]  = (_Float16)a;
        Wsum16[i] = (_Float16)(a + b);
    }
}

// ---------------------------------------------------------------------------
// One LSTM step, fully fused: gates = Xin @ W16^T + bias ; cell update ;
// writes h (f32) into concat[m*8+t, :] and h (f16) into Xout.
// Grid: 128 blocks (one per 16-wide gate-column tile), 256 threads (8 waves).
// Wave w owns rows [32w, 32w+32) = two 16x16 M-tiles and all 4 gate tiles of
// its column, so the LSTM cell nonlinearity fuses on the WMMA accumulators.
//
// Pipeline: double-buffered LDS B tile (2 x 8KB) filled with
// global_load_async_to_lds_b128; K-loop manually unrolled by 2 so the A
// operands alternate between two named register sets (no v_mov ping-pong) and
// the staging calls are unconditional except for the final chunk.
// ---------------------------------------------------------------------------
__global__ __launch_bounds__(256) void lstm_step_kernel(
    const _Float16* __restrict__ W16,   // [8192, 2048] row-major f16
    const float*    __restrict__ bias,  // [8192]
    const _Float16* __restrict__ Xin,   // [256, 2048] f16 (x == h)
    _Float16*       __restrict__ Xout,  // [256, 2048] f16 (next h)
    float*          __restrict__ C,     // [256, 2048] cell state
    float*          __restrict__ concat,// [2048, 2048]
    int t)
{
    __shared__ __align__(64) _Float16 Bs[2][64 * KCH];   // 2 x (4 gates x 16 cols x 64 K)

    const int tid  = threadIdx.x;
    const int lane = tid & 31;
    const int wave = tid >> 5;
    const int n0   = blockIdx.x * 16;   // column tile within each gate's 2048
    const int m0   = wave * 32;         // two M-tiles per wave
    const int col  = lane & 15;
    const int hi16 = lane >> 4;         // lane group 0/1

    v8f acc[2][4];
    const v8f vzero = {};
#pragma unroll
    for (int mt = 0; mt < 2; ++mt)
#pragma unroll
        for (int g = 0; g < 4; ++g) acc[mt][g] = vzero;

    // B staging: row rB in [0,64): gate = rB>>4, col = rB&15 ; each thread
    // moves 2 x 16B per chunk (4 gate tiles = 8KB per K-chunk).
    const int rB = tid >> 2;
    const int c8 = (tid & 3) * 8;
    const _Float16* wrow = W16 + (size_t)((rB >> 4) * NOPX + n0 + (rB & 15)) * NOPX + c8;
    _Float16* bsdst[2] = { &Bs[0][rB * KCH + c8], &Bs[1][rB * KCH + c8] };

    auto stageB = [&](int buf, int kk) {
#if USE_ASYNC_LDS
        __builtin_amdgcn_global_load_async_to_lds_b128(
            (agp_t)(wrow + kk),      (alp_t)(bsdst[buf]),      0, 0);
        __builtin_amdgcn_global_load_async_to_lds_b128(
            (agp_t)(wrow + kk + 32), (alp_t)(bsdst[buf] + 32), 0, 0);
#else
        *(v8h*)(bsdst[buf])      = *(const v8h*)(wrow + kk);
        *(v8h*)(bsdst[buf] + 32) = *(const v8h*)(wrow + kk + 32);
        __builtin_prefetch(wrow + kk + 2 * KCH, 0, 3);
#endif
    };

    // A rows: lane m holds row m; per 32-K slice the K chunks are
    // {0-15} (lanes 0-15) / {16-31} (lanes 16-31), matching the contiguous
    // B K order read from LDS.
    const _Float16* arow0 = Xin + (size_t)(m0 + col) * NOPX + hi16 * 16;
    const _Float16* arow1 = arow0 + (size_t)16 * NOPX;

    auto loadA = [&](const _Float16* base, int off) -> v16h {
        v8h lo = *(const v8h*)(base + off);
        v8h hi = *(const v8h*)(base + off + 8);
        return __builtin_shufflevector(lo, hi,
            0,1,2,3,4,5,6,7,8,9,10,11,12,13,14,15);
    };

    v16h A0[2][2], A1[2][2];   // [mtile][ks], two independent pipeline sets

    auto loadAset = [&](v16h (&A)[2][2], int kk) {
        A[0][0] = loadA(arow0, kk);  A[0][1] = loadA(arow0, kk + 32);
        A[1][0] = loadA(arow1, kk);  A[1][1] = loadA(arow1, kk + 32);
    };

    auto compute = [&](const v16h (&A)[2][2], int buf) {
        const _Float16* bbase = &Bs[buf][col * KCH + hi16 * 16];
#pragma unroll
        for (int ks = 0; ks < 2; ++ks) {
#pragma unroll
            for (int g = 0; g < 4; ++g) {
                v16h b = *(const v16h*)(bbase + g * 16 * KCH + ks * 32);
                acc[0][g] = __builtin_amdgcn_wmma_f32_16x16x32_f16(
                    false, A[0][ks], false, b, (short)0, acc[0][g], false, false);
                acc[1][g] = __builtin_amdgcn_wmma_f32_16x16x32_f16(
                    false, A[1][ks], false, b, (short)0, acc[1][g], false, false);
            }
        }
    };

    // prologue: stage K-chunk 0 into buf 0, load A set 0
    stageB(0, 0);
    loadAset(A0, 0);

    for (int it = 0; it < NIT; it += 2) {
        // even phase (buf 0): it+1 < NIT always (NIT even)
#if USE_ASYNC_LDS
        wait_async_zero();
#endif
        __syncthreads();
        stageB(1, (it + 1) * KCH);
        loadAset(A1, (it + 1) * KCH);
        compute(A0, 0);

        // odd phase (buf 1): stage chunk it+2 unless this is the last pair
#if USE_ASYNC_LDS
        wait_async_zero();
#endif
        __syncthreads();
        if (it + 2 < NIT) {            // uniform scalar branch
            stageB(0, (it + 2) * KCH);
            loadAset(A0, (it + 2) * KCH);
        }
        compute(A1, 1);
    }

    // fused bias + LSTM cell (PyTorch gate order i,f,g,o)
    float bg[4];
#pragma unroll
    for (int g = 0; g < 4; ++g) bg[g] = bias[g * NOPX + n0 + col];

    const int n = n0 + col;
#pragma unroll
    for (int mt = 0; mt < 2; ++mt) {
#pragma unroll
        for (int r = 0; r < 8; ++r) {
            int m = m0 + mt * 16 + hi16 * 8 + r;     // C/D layout: M = r + 8*hi16
            float vi = acc[mt][0][r] + bg[0];
            float vf = acc[mt][1][r] + bg[1];
            float vg = acc[mt][2][r] + bg[2];
            float vo = acc[mt][3][r] + bg[3];
            size_t cidx = (size_t)m * NOPX + n;
            float cold = C[cidx];
            float cn = fsig(vf) * cold + fsig(vi) * ftanh(vg);
            float h  = fsig(vo) * ftanh(cn);
            C[cidx] = cn;
            concat[((size_t)m * MAX_ORD + t) * NOPX + n] = h;
            Xout[cidx] = (_Float16)h;
        }
    }
}

// ---------------------------------------------------------------------------
// softmax (in place, per row) + per-row entropy partial (fixed order)
// ---------------------------------------------------------------------------
__global__ __launch_bounds__(256) void softmax_entropy_kernel(
    float* __restrict__ P, float* __restrict__ entbuf)
{
    __shared__ float red[256];
    const int tid = threadIdx.x;
    float* row = P + (size_t)blockIdx.x * NOPX;

    float mx = -3.4e38f;
    for (int i = tid; i < NOPX; i += 256) mx = fmaxf(mx, row[i]);
    red[tid] = mx; __syncthreads();
    for (int s = 128; s > 0; s >>= 1) {
        if (tid < s) red[tid] = fmaxf(red[tid], red[tid + s]);
        __syncthreads();
    }
    mx = red[0]; __syncthreads();

    float sum = 0.0f;
    for (int i = tid; i < NOPX; i += 256) sum += __expf(row[i] - mx);
    red[tid] = sum; __syncthreads();
    for (int s = 128; s > 0; s >>= 1) {
        if (tid < s) red[tid] += red[tid + s];
        __syncthreads();
    }
    float logZ = __logf(red[0]); __syncthreads();

    float ent = 0.0f;
    for (int i = tid; i < NOPX; i += 256) {
        float lp = row[i] - mx - logZ;
        float pr = __expf(lp);
        row[i] = pr;
        ent -= pr * lp;
    }
    red[tid] = ent; __syncthreads();
    for (int s = 128; s > 0; s >>= 1) {
        if (tid < s) red[tid] += red[tid + s];
        __syncthreads();
    }
    if (tid == 0) entbuf[blockIdx.x] = red[0];
}

// ---------------------------------------------------------------------------
// pick loss: sum(-log(P.flat[index]) * rewards), block partials (grid = 256)
// ---------------------------------------------------------------------------
__global__ __launch_bounds__(256) void pick_loss_kernel(
    const float* __restrict__ P, const int* __restrict__ index,
    const float* __restrict__ rewards, float* __restrict__ pickbuf)
{
    __shared__ float red[256];
    const int i = blockIdx.x * 256 + threadIdx.x;   // 65536 total
    float v = 0.0f;
    if (i < NBATCH * NACT) {
        int idx = index[i];
        v = -__logf(P[idx]) * rewards[i];
    }
    red[threadIdx.x] = v; __syncthreads();
    for (int s = 128; s > 0; s >>= 1) {
        if (threadIdx.x < s) red[threadIdx.x] += red[threadIdx.x + s];
        __syncthreads();
    }
    if (threadIdx.x == 0) pickbuf[blockIdx.x] = red[0];
}

// ---------------------------------------------------------------------------
// finalize: deterministic single-block reduction of all partials
// ---------------------------------------------------------------------------
__global__ __launch_bounds__(256) void finalize_kernel(
    const float* __restrict__ loss_init, const float* __restrict__ entbuf,
    const float* __restrict__ regbuf, const float* __restrict__ pickbuf,
    float* __restrict__ out)
{
    __shared__ float red[256];
    const int tid = threadIdx.x;
    float E = 0.0f, R = 0.0f, PK = 0.0f;

    float a = 0.0f;
    for (int i = tid; i < 2048; i += 256) a += entbuf[i];
    red[tid] = a; __syncthreads();
    for (int s = 128; s > 0; s >>= 1) { if (tid < s) red[tid] += red[tid + s]; __syncthreads(); }
    if (tid == 0) E = red[0];
    __syncthreads();

    a = 0.0f;
    for (int i = tid; i < 512; i += 256) a += regbuf[i];
    red[tid] = a; __syncthreads();
    for (int s = 128; s > 0; s >>= 1) { if (tid < s) red[tid] += red[tid + s]; __syncthreads(); }
    if (tid == 0) R = red[0];
    __syncthreads();

    red[tid] = pickbuf[tid]; __syncthreads();
    for (int s = 128; s > 0; s >>= 1) { if (tid < s) red[tid] += red[tid + s]; __syncthreads(); }
    if (tid == 0) {
        PK = red[0];
        out[0] = (loss_init[0] + PK + (float)NBATCH * (REGC * E + REGC * R)) / (float)NBATCH;
    }
}

// ---------------------------------------------------------------------------
// host launcher
// inputs (dict order): actions, rewards, loss_init, input0, W_ih, W_hh,
//                      b_ih, b_hh, index
// ---------------------------------------------------------------------------
extern "C" void kernel_launch(void* const* d_in, const int* in_sizes, int n_in,
                              void* d_out, int out_size, void* d_ws, size_t ws_size,
                              hipStream_t stream)
{
    (void)in_sizes; (void)n_in; (void)out_size; (void)ws_size;
    // const float* actions = (const float*)d_in[0];  // unused by the reference
    const float* rewards   = (const float*)d_in[1];
    const float* loss_init = (const float*)d_in[2];
    const float* input0    = (const float*)d_in[3];
    const float* W_ih      = (const float*)d_in[4];
    const float* W_hh      = (const float*)d_in[5];
    const float* b_ih      = (const float*)d_in[6];
    const float* b_hh      = (const float*)d_in[7];
    const int*   index     = (const int*)  d_in[8];
    float* out = (float*)d_out;

    char* ws = (char*)d_ws;
    size_t o = 0;
    auto alloc = [&](size_t bytes) -> char* {
        char* r = ws + o;
        o = (o + bytes + 255) & ~(size_t)255;
        return r;
    };
    float*    entbuf  = (float*)   alloc(2048 * sizeof(float));
    float*    regbuf  = (float*)   alloc(512 * sizeof(float));
    float*    pickbuf = (float*)   alloc(256 * sizeof(float));
    float*    bias    = (float*)   alloc((size_t)NG4 * sizeof(float));
    _Float16* xa      = (_Float16*)alloc((size_t)NF * NOPX * sizeof(_Float16));
    _Float16* xb      = (_Float16*)alloc((size_t)NF * NOPX * sizeof(_Float16));
    float*    C       = (float*)   alloc((size_t)NF * NOPX * sizeof(float));
    _Float16* Wih16   = (_Float16*)alloc((size_t)NG4 * NOPX * sizeof(_Float16));
    _Float16* Wsum16  = (_Float16*)alloc((size_t)NG4 * NOPX * sizeof(_Float16));
    float*    concat  = (float*)   alloc((size_t)NACT * NOPX * sizeof(float));
    // total ~88 MB of workspace

    prep_kernel<<<512, 256, 0, stream>>>(input0, W_ih, b_ih, b_hh, bias, xa, C, regbuf);
    convert_kernel<<<4096, 256, 0, stream>>>(W_ih, W_hh, Wih16, Wsum16);

    _Float16* xin = xa;
    _Float16* xout = xb;
    for (int t = 0; t < MAX_ORD; ++t) {
        const _Float16* W = (t == 0) ? Wih16 : Wsum16;  // x==h for t>=1
        lstm_step_kernel<<<128, 256, 0, stream>>>(W, bias, xin, xout, C, concat, t);
        _Float16* tmp = xin; xin = xout; xout = tmp;
    }

    softmax_entropy_kernel<<<NACT, 256, 0, stream>>>(concat, entbuf);
    pick_loss_kernel<<<256, 256, 0, stream>>>(concat, index, rewards, pickbuf);
    finalize_kernel<<<1, 256, 0, stream>>>(loss_init, entbuf, regbuf, pickbuf, out);
}
